// SelfAttentionHead_65420941853169
// MI455X (gfx1250) — matrix operand order repaired and checked by reading, compiled, not run on hardware
//
#include <hip/hip_runtime.h>

#define DEVI __device__ __forceinline__

typedef unsigned short u16;
typedef __attribute__((ext_vector_type(16))) __bf16 v16bf;
typedef __attribute__((ext_vector_type(8)))  __bf16 v8bfe;
typedef __attribute__((ext_vector_type(8)))  float  v8f;
typedef __attribute__((ext_vector_type(4)))  float  f32x4;
typedef __attribute__((ext_vector_type(8)))  float  f32x8;
typedef __attribute__((ext_vector_type(16))) u16    u16x16;
typedef __attribute__((ext_vector_type(8)))  u16    u16x8;
typedef __attribute__((ext_vector_type(4)))  unsigned int u32x4;
typedef __attribute__((ext_vector_type(8)))  int    i32x8;
typedef __attribute__((ext_vector_type(4)))  int    i32x4;

constexpr int B = 4, T = 4096, C = 1024, D = 64;

#if defined(__has_builtin)
#if __has_builtin(__builtin_amdgcn_tensor_load_to_lds) && __has_builtin(__builtin_amdgcn_s_wait_tensorcnt)
#define USE_TDM 1
#endif
#endif

DEVI u16 bfbits(float f) { return __builtin_bit_cast(u16, (__bf16)f); }

DEVI u16x8 cvt8(f32x8 f) {
  return __builtin_bit_cast(u16x8, __builtin_convertvector(f, v8bfe));
}

DEVI v16bf join16(u16x8 lo, u16x8 hi) {
  u16x16 u = __builtin_shufflevector(lo, hi, 0,1,2,3,4,5,6,7,8,9,10,11,12,13,14,15);
  return __builtin_bit_cast(v16bf, u);
}

DEVI f32x8 joinf8(f32x4 lo, f32x4 hi) {
  return __builtin_shufflevector(lo, hi, 0,1,2,3,4,5,6,7);
}

// RA = reuse-A hint: set when previous WMMA (same opcode) used the same A operand.
template <bool RA>
DEVI v8f wmma_bf16(v16bf a, v16bf b, v8f c) {
  return __builtin_amdgcn_wmma_f32_16x16x32_bf16(false, a, false, b, (short)0, c, RA, false);
}

#ifdef USE_TDM
// Issue TDM copy of one contiguous (nelem8 * 8B) chunk to LDS (1-D tile, 8B elements).
DEVI void tdm_load_k(const u16* gptr, unsigned ldsOff, unsigned nelem8) {
  unsigned long long ga = (unsigned long long)(size_t)gptr;
  u32x4 g0;
  g0[0] = 1u;                                                 // count=1, user descriptor
  g0[1] = ldsOff;                                             // lds_addr (bytes)
  g0[2] = (unsigned)ga;                                       // global_addr[31:0]
  g0[3] = ((unsigned)(ga >> 32) & 0x01FFFFFFu) | 0x80000000u; // addr[56:32] | type=2
  i32x8 g1;
  g1[0] = (int)(3u << 16);          // data_size = 3 (8B)
  g1[1] = (int)(nelem8 << 16);      // tensor_dim0
  g1[2] = (int)(1u << 16);          // tensor_dim1 = 1
  g1[3] = (int)(nelem8 << 16);      // tile_dim0
  g1[4] = 0;                        // tile_dim1/2 unused
  g1[5] = (int)nelem8;              // tensor_dim0_stride
  g1[6] = 0; g1[7] = 0;
  i32x4 gz4 = {0, 0, 0, 0};
  i32x8 gz8 = {0, 0, 0, 0, 0, 0, 0, 0};
  __builtin_amdgcn_tensor_load_to_lds(g0, g1, gz4, gz4, gz8, 0);
}
#endif

// ------------- Kernel 0: one-time W fp32 -> bf16, pre-transposed Wt[n][k] -------------
__global__ __launch_bounds__(256) void conv_wt(
    const float* __restrict__ Wq, const float* __restrict__ Wk, const float* __restrict__ Wv,
    u16* __restrict__ tq, u16* __restrict__ tk, u16* __restrict__ tv)
{
  int idx = blockIdx.x * 256 + threadIdx.x;   // 0 .. C*D-1
  int k = idx >> 6, n = idx & 63;             // source W[k][n] row-major [C][D]
  size_t s = (size_t)k * D + n;
  size_t d = (size_t)n * C + k;               // dest Wt[n][k]  [D][C]
  tq[d] = bfbits(Wq[s]);
  tk[d] = bfbits(Wk[s]);
  tv[d] = bfbits(Wv[s]);
}

// ------------- Kernel 1: fused QKV projection, LDS-free -------------
__global__ __launch_bounds__(256) void qkv_proj(
    const float* __restrict__ x,
    const u16* __restrict__ wtq, const u16* __restrict__ wtk, const u16* __restrict__ wtv,
    u16* __restrict__ qb, u16* __restrict__ kb, u16* __restrict__ vb)
{
  const int tid = threadIdx.x;
  const int tBase = blockIdx.x * 128;
  const int w = tid >> 5, l = tid & 31, h = l >> 4, ln = l & 15;

  const int row = tBase + w * 16 + ln;        // A-frag row is private to this lane
  const float* xr = x + (size_t)row * C;

  v8f accq[4] = {}, acck[4] = {}, accv[4] = {};

  for (int kk = 0; kk < C; kk += 32) {
    // A fragment: contiguous K = {16h..16h+15}
    const f32x4* p = (const f32x4*)(xr + kk);
    u16x8 c0 = cvt8(joinf8(p[4 * h],     p[4 * h + 1]));
    u16x8 c1 = cvt8(joinf8(p[4 * h + 2], p[4 * h + 3]));
    v16bf a = join16(c0, c1);

    v16bf bq[4], bk[4], bv[4];
    #pragma unroll
    for (int t = 0; t < 4; ++t) {
      size_t base = (size_t)(t * 16 + ln) * C + kk + 16 * h;   // 32B-aligned
      bq[t] = __builtin_bit_cast(v16bf, *(const u16x16*)(wtq + base));
      bk[t] = __builtin_bit_cast(v16bf, *(const u16x16*)(wtk + base));
      bv[t] = __builtin_bit_cast(v16bf, *(const u16x16*)(wtv + base));
    }
    // All 12 WMMAs share A -> reuse-A hint on all but the first.
    accq[0] = wmma_bf16<false>(a, bq[0], accq[0]);
    acck[0] = wmma_bf16<true >(a, bk[0], acck[0]);
    accv[0] = wmma_bf16<true >(a, bv[0], accv[0]);
    #pragma unroll
    for (int t = 1; t < 4; ++t) {
      accq[t] = wmma_bf16<true>(a, bq[t], accq[t]);
      acck[t] = wmma_bf16<true>(a, bk[t], acck[t]);
      accv[t] = wmma_bf16<true>(a, bv[t], accv[t]);
    }
  }

  #pragma unroll
  for (int t = 0; t < 4; ++t) {
    #pragma unroll
    for (int vv = 0; vv < 8; ++vv) {
      int r = tBase + w * 16 + vv + 8 * h;    // C-layout: M = vv + 8h, N = ln
      size_t idx = (size_t)r * D + t * 16 + ln;
      qb[idx] = bfbits(accq[t][vv]);
      kb[idx] = bfbits(acck[t][vv]);
      vb[idx] = bfbits(accv[t][vv]);
    }
  }
}

// ------------- Kernel 2: causal flash attention, 64-key steps, double-buffered -------------
__global__ __launch_bounds__(256) void flash_attn(
    const u16* __restrict__ qb, const u16* __restrict__ kb, const u16* __restrict__ vb,
    float* __restrict__ out)
{
  __shared__ __align__(16) u16 ks[2][64][64];   // K blocks row-major (double buffered)
  __shared__ __align__(16) u16 vs[2][64][64];   // V blocks transposed [d][key]
  __shared__ __align__(16) u16 ps[8][16][64];   // per-wave P tile

  const int tid = threadIdx.x;
  const int b = blockIdx.y;
  const int qBase = blockIdx.x * 128;
  const int w = tid >> 5, l = tid & 31, h = l >> 4, ln = l & 15;

  const int qrow = b * T + qBase + w * 16 + ln;
  const u16* qp = qb + (size_t)qrow * D;
  v16bf qa0 = join16(*(const u16x8*)(qp +      16 * h), *(const u16x8*)(qp + 8 + 16 * h));
  v16bf qa1 = join16(*(const u16x8*)(qp + 32 + 16 * h), *(const u16x8*)(qp + 40 + 16 * h));

  float m[8], lsum[8];
  v8f o[4] = {};
  #pragma unroll
  for (int i = 0; i < 8; ++i) { m[i] = -1e30f; lsum[i] = 0.f; }

  const int nSteps = (qBase + 128) / 64;       // uniform across block
  const int wmax = qBase + w * 16 + 15;        // last row this wave owns
  const int lr = tid >> 2, lc = (tid & 3) * 16;

#ifdef USE_TDM
  const unsigned ldsK0 = (unsigned)(size_t)&ks[0][0][0];
  const unsigned ldsK1 = (unsigned)(size_t)&ks[1][0][0];
#endif

  // ---- prologue: stage step 0 into buffer 0 ----
  {
    size_t src = (size_t)(b * T + lr) * D + lc;
#ifdef USE_TDM
    if (w == 0) tdm_load_k(kb + (size_t)(b * T) * D, ldsK0, 1024);
#else
    *(u16x8*)&ks[0][lr][lc]     = *(const u16x8*)(kb + src);
    *(u16x8*)&ks[0][lr][lc + 8] = *(const u16x8*)(kb + src + 8);
#endif
    u16x8 pv0 = *(const u16x8*)(vb + src);
    u16x8 pv1 = *(const u16x8*)(vb + src + 8);
    #pragma unroll
    for (int i = 0; i < 8; ++i) { vs[0][lc + i][lr] = pv0[i]; vs[0][lc + 8 + i][lr] = pv1[i]; }
#ifdef USE_TDM
    if (w == 0) __builtin_amdgcn_s_wait_tensorcnt(0);
#endif
  }
  __syncthreads();

  for (int s = 0; s < nSteps; ++s) {
    const int cur = s & 1, nxt = cur ^ 1;
    const int jBase = s * 64;
    const bool pre = (s + 1 < nSteps);

    // ---- prefetch step s+1 (TDM for K; V into registers, scattered after compute) ----
    u16x8 pv0 = {}, pv1 = {};
#ifndef USE_TDM
    u16x8 pk0 = {}, pk1 = {};
#endif
    if (pre) {
      size_t src = (size_t)(b * T + jBase + 64 + lr) * D + lc;
#ifdef USE_TDM
      if (w == 0) tdm_load_k(kb + (size_t)(b * T + jBase + 64) * D, nxt ? ldsK1 : ldsK0, 1024);
#else
      pk0 = *(const u16x8*)(kb + src);
      pk1 = *(const u16x8*)(kb + src + 8);
#endif
      pv0 = *(const u16x8*)(vb + src);
      pv1 = *(const u16x8*)(vb + src + 8);
    }

    // ---- compute on current buffer (wave-uniform skip of fully-masked blocks) ----
    if (jBase <= wmax) {
      // S = Q K^T : 4 column tiles of 16, K-dim 64 = 2 WMMAs each
      v8f sa[4];
      #pragma unroll
      for (int ct = 0; ct < 4; ++ct) {
        v8f z = {};
        v16bf kf = __builtin_bit_cast(v16bf, *(const u16x16*)&ks[cur][ct * 16 + ln][16 * h]);
        sa[ct] = (ct == 0) ? wmma_bf16<false>(qa0, kf, z) : wmma_bf16<true>(qa0, kf, z);
      }
      #pragma unroll
      for (int ct = 0; ct < 4; ++ct) {
        v16bf kf = __builtin_bit_cast(v16bf, *(const u16x16*)&ks[cur][ct * 16 + ln][32 + 16 * h]);
        sa[ct] = (ct == 0) ? wmma_bf16<false>(qa1, kf, sa[ct]) : wmma_bf16<true>(qa1, kf, sa[ct]);
      }

      #pragma unroll
      for (int vv = 0; vv < 8; ++vv) {
        int rowq = qBase + w * 16 + vv + 8 * h;
        float e0 = (jBase      + ln <= rowq) ? sa[0][vv] * 0.25f : -1e30f;
        float e1 = (jBase + 16 + ln <= rowq) ? sa[1][vv] * 0.25f : -1e30f;
        float e2 = (jBase + 32 + ln <= rowq) ? sa[2][vv] * 0.25f : -1e30f;
        float e3 = (jBase + 48 + ln <= rowq) ? sa[3][vv] * 0.25f : -1e30f;
        float tmax = fmaxf(fmaxf(e0, e1), fmaxf(e2, e3));
        tmax = fmaxf(tmax, __shfl_xor(tmax, 1, 32));
        tmax = fmaxf(tmax, __shfl_xor(tmax, 2, 32));
        tmax = fmaxf(tmax, __shfl_xor(tmax, 4, 32));
        tmax = fmaxf(tmax, __shfl_xor(tmax, 8, 32));
        float mn   = fmaxf(m[vv], tmax);
        float corr = __expf(m[vv] - mn);
        float p0 = __expf(e0 - mn), p1 = __expf(e1 - mn);
        float p2 = __expf(e2 - mn), p3 = __expf(e3 - mn);
        float rs = (p0 + p1) + (p2 + p3);
        rs += __shfl_xor(rs, 1, 32);
        rs += __shfl_xor(rs, 2, 32);
        rs += __shfl_xor(rs, 4, 32);
        rs += __shfl_xor(rs, 8, 32);
        lsum[vv] = lsum[vv] * corr + rs;
        m[vv] = mn;
        o[0][vv] *= corr; o[1][vv] *= corr; o[2][vv] *= corr; o[3][vv] *= corr;
        int rl = vv + 8 * h;
        ps[w][rl][ln]      = bfbits(p0);
        ps[w][rl][16 + ln] = bfbits(p1);
        ps[w][rl][32 + ln] = bfbits(p2);
        ps[w][rl][48 + ln] = bfbits(p3);
      }
      asm volatile("s_wait_dscnt 0" ::: "memory");  // same-wave DS in-order
      __builtin_amdgcn_wave_barrier();

      // P as two A-fragments (keys 0..31 and 32..63), then O += P * V
      v16bf pa0 = join16(*(const u16x8*)&ps[w][ln][16 * h],
                         *(const u16x8*)&ps[w][ln][8 + 16 * h]);
      v16bf pa1 = join16(*(const u16x8*)&ps[w][ln][32 + 16 * h],
                         *(const u16x8*)&ps[w][ln][40 + 16 * h]);
      #pragma unroll
      for (int t = 0; t < 4; ++t) {
        v16bf vf = __builtin_bit_cast(v16bf, *(const u16x16*)&vs[cur][t * 16 + ln][16 * h]);
        o[t] = (t == 0) ? wmma_bf16<false>(pa0, vf, o[t]) : wmma_bf16<true>(pa0, vf, o[t]);
      }
      #pragma unroll
      for (int t = 0; t < 4; ++t) {
        v16bf vf = __builtin_bit_cast(v16bf, *(const u16x16*)&vs[cur][t * 16 + ln][32 + 16 * h]);
        o[t] = (t == 0) ? wmma_bf16<false>(pa1, vf, o[t]) : wmma_bf16<true>(pa1, vf, o[t]);
      }
    }

    // ---- commit prefetched data to the next buffer ----
    if (pre) {
#ifndef USE_TDM
      *(u16x8*)&ks[nxt][lr][lc]     = pk0;
      *(u16x8*)&ks[nxt][lr][lc + 8] = pk1;
#endif
      #pragma unroll
      for (int i = 0; i < 8; ++i) { vs[nxt][lc + i][lr] = pv0[i]; vs[nxt][lc + 8 + i][lr] = pv1[i]; }
#ifdef USE_TDM
      if (w == 0) __builtin_amdgcn_s_wait_tensorcnt(0);
#endif
    }
    __syncthreads();
  }

  #pragma unroll
  for (int vv = 0; vv < 8; ++vv) {
    float inv = lsum[vv] > 0.f ? 1.0f / lsum[vv] : 0.f;
    int rowg = b * T + qBase + w * 16 + vv + 8 * h;
    float* op = out + (size_t)rowg * D;
    #pragma unroll
    for (int t = 0; t < 4; ++t) op[t * 16 + ln] = o[t][vv] * inv;
  }
}

extern "C" void kernel_launch(void* const* d_in, const int* in_sizes, int n_in,
                              void* d_out, int out_size, void* d_ws, size_t ws_size,
                              hipStream_t stream) {
  const float* x  = (const float*)d_in[0];
  const float* Wq = (const float*)d_in[1];
  const float* Wk = (const float*)d_in[2];
  const float* Wv = (const float*)d_in[3];

  u16* qb  = (u16*)d_ws;                        // 3 x B*T*D bf16 = 6 MB
  u16* kb  = qb + (size_t)B * T * D;
  u16* vb  = kb + (size_t)B * T * D;
  u16* wtq = vb + (size_t)B * T * D;            // 3 x D*C bf16 = 384 KB, transposed
  u16* wtk = wtq + (size_t)D * C;
  u16* wtv = wtk + (size_t)D * C;

  conv_wt <<<dim3((C * D) / 256), 256, 0, stream>>>(Wq, Wk, Wv, wtq, wtk, wtv);
  qkv_proj<<<dim3((B * T) / 128), 256, 0, stream>>>(x, wtq, wtk, wtv, qb, kb, vb);
  flash_attn<<<dim3(T / 128, B), 256, 0, stream>>>(qb, kb, vb, (float*)d_out);
}
